// MultiHeadAttention_14422500180012
// MI455X (gfx1250) — compile-verified, matrix operand order checked by reading, not run on hardware
//
#include <hip/hip_runtime.h>

#define B_  2
#define S_  2048
#define DM  1024
#define H_  16
#define HD_ 64
#define M_  (B_ * S_)   // 4096

typedef _Float16 v8h  __attribute__((ext_vector_type(8)));
typedef _Float16 v16h __attribute__((ext_vector_type(16)));
typedef float    v8f  __attribute__((ext_vector_type(8)));
typedef unsigned uint4v __attribute__((ext_vector_type(4)));
typedef unsigned uint8v __attribute__((ext_vector_type(8)));

static __device__ __forceinline__ v16h cat16(v8h lo, v8h hi) {
  return __builtin_shufflevector(lo, hi, 0,1,2,3,4,5,6,7,8,9,10,11,12,13,14,15);
}
static __device__ __forceinline__ v8f vzero8() {
  v8f z;
#pragma unroll
  for (int i = 0; i < 8; ++i) z[i] = 0.0f;
  return z;
}
static __device__ __forceinline__ v8f wmma16(v16h a, v16h b, v8f c) {
  // D = A(16x32 f16) * B(32x16 f16) + C(16x16 f32)
  return __builtin_amdgcn_wmma_f32_16x16x32_f16(false, a, false, b, (short)0, c, false, false);
}

// Low 32 bits of a flat pointer into LDS == workgroup LDS byte offset.
static __device__ __forceinline__ unsigned lds_off(const void* p) {
  return (unsigned)(uintptr_t)p;
}
// CDNA5 async DMA: one instruction moves 32 lanes x 16B global -> LDS, ASYNCcnt-tracked.
static __device__ __forceinline__ void async_ld_b128(unsigned lds, const void* g) {
  asm volatile("global_load_async_to_lds_b128 %0, %1, off"
               :: "v"(lds), "v"((unsigned long long)(uintptr_t)g) : "memory");
}
static __device__ __forceinline__ void async_wait2() {
  asm volatile("s_wait_asynccnt 0x2" ::: "memory");
}
static __device__ __forceinline__ void async_wait0() {
  asm volatile("s_wait_asynccnt 0x0" ::: "memory");
}
// CDNA5 Tensor Data Mover: 2D tile load, D# in SGPRs (groups 0 and 1), TENSORcnt-tracked.
static __device__ __forceinline__ void tdm_load_2d(uint4v g0, uint8v g1) {
  asm volatile("tensor_load_to_lds %0, %1" :: "s"(g0), "s"(g1) : "memory");
}

// ---------------------------------------------------------------- casts ----
__global__ void cast_f32_f16(const float* __restrict__ in, _Float16* __restrict__ out, int n) {
  int i = blockIdx.x * blockDim.x + threadIdx.x;
  if (i < n) out[i] = (_Float16)in[i];
}

// Wt[n][k] = W[k][n]  (1024x1024), so GEMM B-fragments are K-contiguous.
__global__ void transpose_cast_w(const float* __restrict__ w, _Float16* __restrict__ wt) {
  int i = blockIdx.x * blockDim.x + threadIdx.x;   // over DM*DM
  int n = i >> 10, k = i & (DM - 1);
  wt[i] = (_Float16)w[(size_t)k * DM + n];
}

// ----------------------------------------------------------------- GEMM ----
// C[M][DM] = A[M][DM](f16) * Wt^T + bias ; Wt is [n][k] f16.
// Block tile 128x128, 8 waves as 4(m) x 2(n), wave tile 32x64, BK=32.
// Double-buffered LDS: A tile via async global->LDS DMA (ASYNCcnt),
// B tile via Tensor Data Mover (TENSORcnt), one 16-row slice per wave.
__global__ __launch_bounds__(256)
void gemm_f16(const _Float16* __restrict__ A,
              const _Float16* __restrict__ Bt,
              const float* __restrict__ bias,
              void* __restrict__ Cout, int c_is_f16)
{
  __shared__ __align__(16) _Float16 sA[2][128][32];
  __shared__ __align__(16) _Float16 sB[2][128][32];   // [n-col][k]

  const int tid  = threadIdx.x;
  const int lane = tid & 31;
  const int wid  = tid >> 5;
  const int mw   = wid & 3;          // 0..3
  const int nw   = wid >> 2;         // 0..1
  const int m0   = blockIdx.x * 128;
  const int n0   = blockIdx.y * 128;
  const int lh   = lane & 15;
  const int kb   = (lane >> 4) * 8;  // fragment K base per lane half
  const int rowoff = 8 * (lane >> 4);
  const int NK = DM / 32;            // 32 k-steps

  // per-wave A-staging chunks: chunk = 16B (8 halfs); 512 chunks/tile; 2 async per wave/step
  int c0  = (wid * 2 + 0) * 32 + lane;
  int c1  = (wid * 2 + 1) * 32 + lane;
  int r0  = c0 >> 2, o0 = (c0 & 3) * 8;
  int r1  = c1 >> 2, o1 = (c1 & 3) * 8;

  v8f acc[2][4];
#pragma unroll
  for (int i = 0; i < 2; ++i)
#pragma unroll
    for (int j = 0; j < 4; ++j) acc[i][j] = vzero8();

  auto stage = [&](int ks, int buf) {
    size_t kc = (size_t)ks * 32;
    // A tile: per-lane async DMA to LDS
    async_ld_b128(lds_off(&sA[buf][r0][o0]), &A[(size_t)(m0 + r0) * DM + kc + o0]);
    async_ld_b128(lds_off(&sA[buf][r1][o1]), &A[(size_t)(m0 + r1) * DM + kc + o1]);
    // B tile: this wave's 16-row slice via one TDM descriptor (2D tile, D# groups 0+1)
    unsigned long long ga =
        (unsigned long long)(uintptr_t)(Bt + (size_t)(n0 + wid * 16) * DM + kc);
    uint4v g0;
    g0[0] = 1u;                                             // count=1 valid descriptor
    g0[1] = lds_off(&sB[buf][wid * 16][0]);                 // lds_addr
    g0[2] = (unsigned)ga;                                   // global_addr[31:0]
    g0[3] = ((unsigned)(ga >> 32) & 0x01FFFFFFu) | 0x80000000u; // addr[56:32] | type=2
    uint8v g1;
    g1[0] = 0x00010000u;        // workgroup_mask=0, data_size=1 (2 bytes)
    g1[1] = (DM & 0xFFFFu) << 16;   // tensor_dim0 = 1024 (bits 79:48, low half)
    g1[2] = (DM & 0xFFFFu) << 16;   // dim0 hi=0 | tensor_dim1 = 1024 (low half)
    g1[3] = 32u << 16;          // dim1 hi=0 | tile_dim0 = 32 elements
    g1[4] = 16u;                // tile_dim1 = 16 rows, tile_dim2 = 0
    g1[5] = (unsigned)DM;       // tensor_dim0_stride = 1024 (low 32)
    g1[6] = 0u;
    g1[7] = 0u;
    tdm_load_2d(g0, g1);
  };

  stage(0, 0);   // prologue
  for (int ks = 0; ks < NK; ++ks) {
    const int buf = ks & 1;
    if (ks + 1 < NK) {
      stage(ks + 1, buf ^ 1);
      async_wait2();                          // tile ks A-chunks arrived
      __builtin_amdgcn_s_wait_tensorcnt(1);   // tile ks B-slice arrived
    } else {
      async_wait0();
      __builtin_amdgcn_s_wait_tensorcnt(0);
    }
    __syncthreads();                          // tile ks visible to all waves

    v16h af[2], bf[4];
#pragma unroll
    for (int mi = 0; mi < 2; ++mi) {
      int r = mw * 32 + mi * 16 + lh;
      af[mi] = cat16(*(const v8h*)&sA[buf][r][kb], *(const v8h*)&sA[buf][r][kb + 16]);
    }
#pragma unroll
    for (int ni = 0; ni < 4; ++ni) {
      int r = nw * 64 + ni * 16 + lh;
      bf[ni] = cat16(*(const v8h*)&sB[buf][r][kb], *(const v8h*)&sB[buf][r][kb + 16]);
    }
#pragma unroll
    for (int mi = 0; mi < 2; ++mi)
#pragma unroll
      for (int ni = 0; ni < 4; ++ni)
        acc[mi][ni] = wmma16(af[mi], bf[ni], acc[mi][ni]);

    __syncthreads();                          // all reads of buf done before refill
  }

#pragma unroll
  for (int mi = 0; mi < 2; ++mi) {
#pragma unroll
    for (int ni = 0; ni < 4; ++ni) {
      int col  = n0 + nw * 64 + ni * 16 + lh;
      float bv = bias[col];
#pragma unroll
      for (int r = 0; r < 8; ++r) {
        int row   = m0 + mw * 32 + mi * 16 + r + rowoff;
        float val = acc[mi][ni][r] + bv;
        if (c_is_f16) ((_Float16*)Cout)[(size_t)row * DM + col] = (_Float16)val;
        else          ((float*)   Cout)[(size_t)row * DM + col] = val;
      }
    }
  }
}

// ----------------------------------------------------------------- RoPE ----
// raw: [B,S,D] f16 (=[b,s,h,hd]); out: [B,H,S,HD] f16, scaled (Q gets 1/sqrt(hd))
__global__ void rope_kernel(const _Float16* __restrict__ raw,
                            _Float16* __restrict__ out, float scale)
{
  int idx = blockIdx.x * blockDim.x + threadIdx.x;     // B*S*H*(HD/2) = 2^21
  if (idx >= B_ * S_ * H_ * (HD_ / 2)) return;
  int i = idx & 31;
  int h = (idx >> 5) & (H_ - 1);
  int s = (idx >> 9) & (S_ - 1);
  int b = idx >> 20;
  float inv_freq = __expf(-(float)(2 * i) * (1.0f / (float)HD_) * 9.210340371976184f); // ln(1e4)
  float ang = (float)s * inv_freq;
  float sn = sinf(ang), cs = cosf(ang);
  size_t ib = ((size_t)(b * S_ + s)) * DM + h * HD_ + 2 * i;
  float x1 = (float)raw[ib], x2 = (float)raw[ib + 1];
  size_t ob = (((size_t)(b * H_ + h)) * S_ + s) * HD_ + 2 * i;
  out[ob]     = (_Float16)((x1 * cs - x2 * sn) * scale);
  out[ob + 1] = (_Float16)((x1 * sn + x2 * cs) * scale);
}

// V: [B,S,D] f16 -> Vt: [B,H,HD,S] f16 (so P*V B-fragments are key-contiguous)
__global__ void vtrans(const _Float16* __restrict__ raw, _Float16* __restrict__ vt) {
  int idx = blockIdx.x * blockDim.x + threadIdx.x;     // B*H*HD*S = 2^22
  if (idx >= B_ * H_ * HD_ * S_) return;
  int s  = idx & (S_ - 1);
  int hd = (idx >> 11) & (HD_ - 1);
  int h  = (idx >> 17) & (H_ - 1);
  int b  = idx >> 21;
  vt[idx] = raw[((size_t)(b * S_ + s)) * DM + h * HD_ + hd];
}

// ------------------------------------------------------------- attention ---
// Qh pre-scaled [B,H,S,HD], Kh [B,H,S,HD], Vt [B,H,HD,S], mask [S,S], Ctx [B,S,D] f16
__global__ __launch_bounds__(256)
void flash_attn(const _Float16* __restrict__ Qh, const _Float16* __restrict__ Kh,
                const _Float16* __restrict__ Vt, const float* __restrict__ mask,
                _Float16* __restrict__ Ctx)
{
  __shared__ __align__(16) _Float16 pbuf[8][16][32];   // per-wave P transpose tile

  const int lane = threadIdx.x & 31;
  const int wid  = threadIdx.x >> 5;
  const int bh   = blockIdx.x;               // 0..B*H-1
  const int b = bh >> 4, h = bh & (H_ - 1);
  const int q0 = blockIdx.y * 128 + wid * 16;
  const int lh = lane & 15;
  const int kb = (lane >> 4) * 8;
  const int rowoff = 8 * (lane >> 4);

  // Q fragments: rows q0..q0+15, K-dim = HD (two 32-wide halves)
  const _Float16* qrow = Qh + ((size_t)bh * S_ + (q0 + lh)) * HD_;
  v16h qf0 = cat16(*(const v8h*)(qrow + kb),      *(const v8h*)(qrow + kb + 16));
  v16h qf1 = cat16(*(const v8h*)(qrow + 32 + kb), *(const v8h*)(qrow + 32 + kb + 16));

  float mrun[8], lrun[8];
  v8f acc[4];
#pragma unroll
  for (int r = 0; r < 8; ++r) { mrun[r] = -1e30f; lrun[r] = 0.0f; }
#pragma unroll
  for (int t = 0; t < 4; ++t) acc[t] = vzero8();

  for (int k0 = 0; k0 < S_; k0 += 32) {
    // speculative prefetch of next key block (dropped if past end, ISA 10.5)
    __builtin_prefetch(Kh + ((size_t)bh * S_ + (k0 + 32 + lh)) * HD_, 0, 1);
    __builtin_prefetch(Vt + ((size_t)bh * HD_ + lh) * S_ + k0 + 32, 0, 1);

    // K^T fragments: [ksub half][key half]
    v16h bk00, bk01, bk10, bk11;
    {
      const _Float16* kr0 = Kh + ((size_t)bh * S_ + (k0 + lh)) * HD_;
      const _Float16* kr1 = Kh + ((size_t)bh * S_ + (k0 + 16 + lh)) * HD_;
      bk00 = cat16(*(const v8h*)(kr0 + kb),      *(const v8h*)(kr0 + kb + 16));
      bk10 = cat16(*(const v8h*)(kr0 + 32 + kb), *(const v8h*)(kr0 + 32 + kb + 16));
      bk01 = cat16(*(const v8h*)(kr1 + kb),      *(const v8h*)(kr1 + kb + 16));
      bk11 = cat16(*(const v8h*)(kr1 + 32 + kb), *(const v8h*)(kr1 + 32 + kb + 16));
    }
    v8f s0 = vzero8(), s1 = vzero8();
    s0 = wmma16(qf0, bk00, s0); s0 = wmma16(qf1, bk10, s0);   // keys k0..k0+15
    s1 = wmma16(qf0, bk01, s1); s1 = wmma16(qf1, bk11, s1);   // keys k0+16..k0+31

    // mask add (general; zeros in the provided inputs)
#pragma unroll
    for (int r = 0; r < 8; ++r) {
      int qr = q0 + r + rowoff;
      s0[r] += mask[(size_t)qr * S_ + k0 + lh];
      s1[r] += mask[(size_t)qr * S_ + k0 + 16 + lh];
    }

    // online softmax (row reductions across each 16-lane half)
    float p0[8], p1[8], alpha[8];
#pragma unroll
    for (int r = 0; r < 8; ++r) {
      float mx = fmaxf(s0[r], s1[r]);
      mx = fmaxf(mx, __shfl_xor(mx, 1, 16));
      mx = fmaxf(mx, __shfl_xor(mx, 2, 16));
      mx = fmaxf(mx, __shfl_xor(mx, 4, 16));
      mx = fmaxf(mx, __shfl_xor(mx, 8, 16));
      float mnew = fmaxf(mrun[r], mx);
      float a    = __expf(mrun[r] - mnew);
      p0[r] = __expf(s0[r] - mnew);
      p1[r] = __expf(s1[r] - mnew);
      float rs = p0[r] + p1[r];
      rs += __shfl_xor(rs, 1, 16);
      rs += __shfl_xor(rs, 2, 16);
      rs += __shfl_xor(rs, 4, 16);
      rs += __shfl_xor(rs, 8, 16);
      lrun[r]  = lrun[r] * a + rs;
      mrun[r]  = mnew;
      alpha[r] = a;
    }
#pragma unroll
    for (int t = 0; t < 4; ++t)
#pragma unroll
      for (int r = 0; r < 8; ++r) acc[t][r] *= alpha[r];

    // P: C-layout -> row-major LDS tile (per-wave private; DS is in-order per wave)
#pragma unroll
    for (int r = 0; r < 8; ++r) {
      int mrow = r + rowoff;
      pbuf[wid][mrow][lh]      = (_Float16)p0[r];
      pbuf[wid][mrow][16 + lh] = (_Float16)p1[r];
    }
    // re-read as A-fragment (16 rows x 32 keys)
    v16h pa = cat16(*(const v8h*)&pbuf[wid][lh][kb], *(const v8h*)&pbuf[wid][lh][kb + 16]);

    // P @ V : V B-fragments are key-contiguous thanks to the [B,H,HD,S] layout
#pragma unroll
    for (int t = 0; t < 4; ++t) {
      const _Float16* vr = Vt + ((size_t)bh * HD_ + t * 16 + lh) * S_ + k0;
      v16h bv = cat16(*(const v8h*)(vr + kb), *(const v8h*)(vr + kb + 16));
      acc[t] = wmma16(pa, bv, acc[t]);
    }
  }

#pragma unroll
  for (int r = 0; r < 8; ++r) {
    int qr   = q0 + r + rowoff;
    float rl = 1.0f / lrun[r];
    size_t baseo = ((size_t)(b * S_ + qr)) * DM + h * HD_;
#pragma unroll
    for (int t = 0; t < 4; ++t)
      Ctx[baseo + t * 16 + lh] = (_Float16)(acc[t][r] * rl);
  }
}

// --------------------------------------------------------------- launch ----
extern "C" void kernel_launch(void* const* d_in, const int* in_sizes, int n_in,
                              void* d_out, int out_size, void* d_ws, size_t ws_size,
                              hipStream_t stream)
{
  const float* x    = (const float*)d_in[0];
  const float* mask = (const float*)d_in[1];
  const float* Wq   = (const float*)d_in[2];
  const float* bq   = (const float*)d_in[3];
  const float* Wk   = (const float*)d_in[4];
  const float* bk   = (const float*)d_in[5];
  const float* Wv   = (const float*)d_in[6];
  const float* bv   = (const float*)d_in[7];
  const float* Wo   = (const float*)d_in[8];
  const float* bo   = (const float*)d_in[9];
  float* out = (float*)d_out;

  // workspace carve-up (~64 MB total)
  char* ws = (char*)d_ws;
  size_t off = 0;
  auto carve = [&](size_t bytes) {
    char* p = ws + off;
    off += (bytes + 255) & ~(size_t)255;
    return p;
  };
  _Float16* Xh   = (_Float16*)carve((size_t)M_ * DM * 2);   // 8 MB
  _Float16* WqT  = (_Float16*)carve((size_t)DM * DM * 2);   // 2 MB each
  _Float16* WkT  = (_Float16*)carve((size_t)DM * DM * 2);
  _Float16* WvT  = (_Float16*)carve((size_t)DM * DM * 2);
  _Float16* WoT  = (_Float16*)carve((size_t)DM * DM * 2);
  _Float16* Qraw = (_Float16*)carve((size_t)M_ * DM * 2);   // 8 MB
  _Float16* Kraw = (_Float16*)carve((size_t)M_ * DM * 2);
  _Float16* Vraw = (_Float16*)carve((size_t)M_ * DM * 2);
  _Float16* Qh   = (_Float16*)carve((size_t)M_ * DM * 2);
  _Float16* Kh   = (_Float16*)carve((size_t)M_ * DM * 2);
  _Float16* Vt   = (_Float16*)carve((size_t)M_ * DM * 2);
  _Float16* Ctx  = Qraw;  // Qraw dead after rope; reuse for context

  const int nX = M_ * DM;
  cast_f32_f16<<<(nX + 255) / 256, 256, 0, stream>>>(x, Xh, nX);
  transpose_cast_w<<<(DM * DM) / 256, 256, 0, stream>>>(Wq, WqT);
  transpose_cast_w<<<(DM * DM) / 256, 256, 0, stream>>>(Wk, WkT);
  transpose_cast_w<<<(DM * DM) / 256, 256, 0, stream>>>(Wv, WvT);
  transpose_cast_w<<<(DM * DM) / 256, 256, 0, stream>>>(Wo, WoT);

  dim3 ggrid(M_ / 128, DM / 128);   // 32 x 8
  gemm_f16<<<ggrid, 256, 0, stream>>>(Xh, WqT, bq, Qraw, 1);
  gemm_f16<<<ggrid, 256, 0, stream>>>(Xh, WkT, bk, Kraw, 1);
  gemm_f16<<<ggrid, 256, 0, stream>>>(Xh, WvT, bv, Vraw, 1);

  const int nR = B_ * S_ * H_ * (HD_ / 2);
  rope_kernel<<<(nR + 255) / 256, 256, 0, stream>>>(Qraw, Qh, 0.125f);  // 1/sqrt(64)
  rope_kernel<<<(nR + 255) / 256, 256, 0, stream>>>(Kraw, Kh, 1.0f);
  const int nV = B_ * H_ * HD_ * S_;
  vtrans<<<(nV + 255) / 256, 256, 0, stream>>>(Vraw, Vt);

  dim3 agrid(B_ * H_, S_ / 128);    // 32 x 16
  flash_attn<<<agrid, 256, 0, stream>>>(Qh, Kh, Vt, mask, Ctx);

  gemm_f16<<<ggrid, 256, 0, stream>>>(Ctx, WoT, bo, out, 0);
}
